// EvolveGCNModel_64372969832579
// MI455X (gfx1250) — matrix operand order repaired and checked by reading, not compiled, not run on hardware
//
#include <hip/hip_runtime.h>

typedef __attribute__((ext_vector_type(16))) __bf16 v16bf;
typedef __attribute__((ext_vector_type(8)))  float  v8f;

#define HDIM 128
#define SLOPE 0.22916666666666666f  // 11/48

// ---------------- Matrix-GRU weight evolution (128x128, tiny) ----------------
__global__ __launch_bounds__(128)
void gru_zr_kernel(const float* __restrict__ Wprev,
                   const float* __restrict__ gW, const float* __restrict__ gU,
                   const float* __restrict__ gb,
                   float* __restrict__ zbuf, float* __restrict__ rW) {
  const int i = blockIdx.x;   // output row
  const int j = threadIdx.x;  // output col
  const float* gw0 = gW + i * HDIM;
  const float* gu0 = gU + i * HDIM;
  const float* gw1 = gW + HDIM * HDIM + i * HDIM;
  const float* gu1 = gU + HDIM * HDIM + i * HDIM;
  float az = 0.f, ar = 0.f;
  for (int k = 0; k < HDIM; ++k) {
    float w = Wprev[k * HDIM + j];
    az = fmaf(gw0[k] + gu0[k], w, az);
    ar = fmaf(gw1[k] + gu1[k], w, ar);
  }
  float z = 1.f / (1.f + expf(-(az + gb[i * HDIM + j])));
  float r = 1.f / (1.f + expf(-(ar + gb[HDIM * HDIM + i * HDIM + j])));
  zbuf[i * HDIM + j] = z;
  rW[i * HDIM + j] = r * Wprev[i * HDIM + j];
}

__global__ __launch_bounds__(128)
void gru_h_kernel(const float* __restrict__ Wprev,
                  const float* __restrict__ gW, const float* __restrict__ gU,
                  const float* __restrict__ gb,
                  const float* __restrict__ zbuf, const float* __restrict__ rW,
                  float* __restrict__ Wnew) {
  const int i = blockIdx.x, j = threadIdx.x;
  const float* gw2 = gW + 2 * HDIM * HDIM + i * HDIM;
  const float* gu2 = gU + 2 * HDIM * HDIM + i * HDIM;
  float ah = 0.f;
  for (int k = 0; k < HDIM; ++k) {
    ah = fmaf(gw2[k], Wprev[k * HDIM + j], ah);
    ah = fmaf(gu2[k], rW[k * HDIM + j], ah);
  }
  float h = tanhf(ah + gb[2 * HDIM * HDIM + i * HDIM + j]);
  float z = zbuf[i * HDIM + j];
  float wp = Wprev[i * HDIM + j];
  Wnew[i * HDIM + j] = (1.f - z) * wp + z * h;
}

// ---------------- XW = X[nrows,128] @ W[128,128] via bf16 WMMA ----------------
// One wave per 16-row tile; 8 N-tiles x 4 K-steps = 32 v_wmma per wave.
// W staged transposed (ldsW[n*128+k]) in LDS as bf16 so B fragments are
// 16 contiguous bf16 per lane.
__global__ __launch_bounds__(256)
void gemm_xw_kernel(const float* __restrict__ X, const float* __restrict__ W,
                    float* __restrict__ XW, int nrows) {
  __shared__ __bf16 ldsW[HDIM * HDIM];
  const int tid = threadIdx.x;
  for (int idx = tid; idx < HDIM * HDIM; idx += 256) {
    int k = idx >> 7, n = idx & 127;
    ldsW[n * HDIM + k] = (__bf16)W[idx];  // transpose: [n][k]
  }
  __syncthreads();

  const int wave = tid >> 5;
  const int lane = tid & 31;
  const int tile = blockIdx.x * 8 + wave;      // 16-row tile
  if (tile * 16 >= nrows) return;              // wave-uniform: whole wave exits
  const int half = lane >> 4;                  // 0: lanes 0-15, 1: lanes 16-31
  const int m = lane & 15;

  int arow = tile * 16 + m;
  if (arow >= nrows) arow = nrows - 1;         // clamp loads; stores guarded
  const float* xrow = X + (size_t)arow * HDIM;

  const v8f vzero = {0.f, 0.f, 0.f, 0.f, 0.f, 0.f, 0.f, 0.f};
  v8f acc[8];
#pragma unroll
  for (int nt = 0; nt < 8; ++nt) acc[nt] = vzero;

#pragma unroll
  for (int kk = 0; kk < 4; ++kk) {
    const int kbase = kk * 32;
    // A fragment: contiguous k order, same as B. Lane(half,m):
    //   elements 0..7  <- K = kbase + 16*half + (0..7)
    //   elements 8..15 <- K = kbase + 16*half + 8 + (0..7)
    v16bf a;
    const float* p0 = xrow + kbase + 16 * half;
    const float* p1 = p0 + 8;
#pragma unroll
    for (int q = 0; q < 8; ++q) {
      a[q]     = (__bf16)p0[q];
      a[8 + q] = (__bf16)p1[q];
    }
#pragma unroll
    for (int nt = 0; nt < 8; ++nt) {
      const int n = nt * 16 + m;
      // B fragment: element i <- W[kbase + 16*half + i][n] (contiguous in ldsW)
      const __bf16* bp = &ldsW[n * HDIM + kbase + 16 * half];
      v16bf b;
#pragma unroll
      for (int q = 0; q < 16; ++q) b[q] = bp[q];
      acc[nt] = __builtin_amdgcn_wmma_f32_16x16x32_bf16(
          false, a, false, b, (short)0, acc[nt], false, false);
    }
  }

  // D layout: VGPR v holds row (v + 8*half), col = nt*16 + m
#pragma unroll
  for (int nt = 0; nt < 8; ++nt) {
#pragma unroll
    for (int v = 0; v < 8; ++v) {
      int row = tile * 16 + v + 8 * half;
      if (row < nrows) XW[(size_t)row * HDIM + nt * 16 + m] = acc[nt][v];
    }
  }
}

// ---------------- SpMM: emb[row] += val * XW[col] (per-edge scatter) --------
__global__ __launch_bounds__(256)
void spmm_scatter_kernel(const int* __restrict__ row, const int* __restrict__ col,
                         const float* __restrict__ val, const float* __restrict__ XW,
                         float* __restrict__ emb, int E) {
  const int e = blockIdx.x * 8 + (threadIdx.x >> 5);  // one edge per wave
  if (e >= E) return;
  const int lane = threadIdx.x & 31;
  const int r = row[e];
  const int c = col[e];
  const float v = val[e];
  const float4 x = *(const float4*)(XW + (size_t)c * HDIM + lane * 4);
  float* dst = emb + (size_t)r * HDIM + lane * 4;
  unsafeAtomicAdd(dst + 0, v * x.x);
  unsafeAtomicAdd(dst + 1, v * x.y);
  unsafeAtomicAdd(dst + 2, v * x.z);
  unsafeAtomicAdd(dst + 3, v * x.w);
}

__global__ __launch_bounds__(256)
void zero_kernel(float* __restrict__ p, int n) {
  int i = blockIdx.x * 256 + threadIdx.x;
  if (i < n) p[i] = 0.f;
}

__global__ __launch_bounds__(256)
void rrelu_kernel(const float* __restrict__ in, float* __restrict__ out, int n) {
  int i = blockIdx.x * 256 + threadIdx.x;
  if (i < n) {
    float x = in[i];
    out[i] = (x >= 0.f) ? x : SLOPE * x;
  }
}

// ---------------- driver ----------------
extern "C" void kernel_launch(void* const* d_in, const int* in_sizes, int n_in,
                              void* d_out, int out_size, void* d_ws, size_t ws_size,
                              hipStream_t stream) {
  const float* features = (const float*)d_in[0];   // [T,N,128] f32
  const int*   adj_row  = (const int*)d_in[1];     // [T,E]
  const int*   adj_col  = (const int*)d_in[2];     // [T,E]
  const float* adj_val  = (const float*)d_in[3];   // [T,E]
  const float* W1  = (const float*)d_in[4];
  const float* g1W = (const float*)d_in[5];
  const float* g1U = (const float*)d_in[6];
  const float* g1b = (const float*)d_in[7];
  const float* W2  = (const float*)d_in[8];
  const float* g2W = (const float*)d_in[9];
  const float* g2U = (const float*)d_in[10];
  const float* g2b = (const float*)d_in[11];
  float* out = (float*)d_out;

  const int T = 4;
  const int E = in_sizes[1] / T;                       // 800000
  const int N = in_sizes[0] / (T * HDIM);              // 50000
  const size_t NH = (size_t)N * HDIM;

  // Workspace: one big buffer + tiny GRU scratch; d_out doubles as 2nd big buf.
  float* bufA  = (float*)d_ws;                 // N*128
  float* zbuf  = bufA + NH;                    // 128*128
  float* rWbuf = zbuf + HDIM * HDIM;
  float* Wb0   = rWbuf + HDIM * HDIM;
  float* Wb1   = Wb0 + HDIM * HDIM;

  const int tiles = (N + 15) / 16;
  const int gemmBlocks = (tiles + 7) / 8;
  const int elemBlocks = (int)((NH + 255) / 256);
  const int edgeBlocks = (E + 7) / 8;
  const size_t adjOff = (size_t)(T - 1) * E;           // only t=3 matters

  // ---- Layer 1: evolve W1 four times (only final weights are ever used) ----
  const float* wp = W1;
  float* wnbuf[2] = {Wb0, Wb1};
  for (int t = 0; t < T; ++t) {
    float* wn = wnbuf[t & 1];
    gru_zr_kernel<<<HDIM, HDIM, 0, stream>>>(wp, g1W, g1U, g1b, zbuf, rWbuf);
    gru_h_kernel<<<HDIM, HDIM, 0, stream>>>(wp, g1W, g1U, g1b, zbuf, rWbuf, wn);
    wp = wn;
  }
  // XW1 = features[3] @ W1_final  -> bufA
  gemm_xw_kernel<<<gemmBlocks, 256, 0, stream>>>(
      features + (size_t)(T - 1) * NH, wp, bufA, N);
  // scatter into d_out (scratch role), rrelu -> bufA (= h1[3])
  zero_kernel<<<elemBlocks, 256, 0, stream>>>(out, (int)NH);
  spmm_scatter_kernel<<<edgeBlocks, 256, 0, stream>>>(
      adj_row + adjOff, adj_col + adjOff, adj_val + adjOff, bufA, out, E);
  rrelu_kernel<<<elemBlocks, 256, 0, stream>>>(out, bufA, (int)NH);

  // ---- Layer 2: evolve W2 four times ----
  wp = W2;
  for (int t = 0; t < T; ++t) {
    float* wn = wnbuf[t & 1];
    gru_zr_kernel<<<HDIM, HDIM, 0, stream>>>(wp, g2W, g2U, g2b, zbuf, rWbuf);
    gru_h_kernel<<<HDIM, HDIM, 0, stream>>>(wp, g2W, g2U, g2b, zbuf, rWbuf, wn);
    wp = wn;
  }
  // XW2 = h1[3] @ W2_final -> d_out (scratch), scatter -> bufA, rrelu -> d_out
  gemm_xw_kernel<<<gemmBlocks, 256, 0, stream>>>(bufA, wp, out, N);
  zero_kernel<<<elemBlocks, 256, 0, stream>>>(bufA, (int)NH);
  spmm_scatter_kernel<<<edgeBlocks, 256, 0, stream>>>(
      adj_row + adjOff, adj_col + adjOff, adj_val + adjOff, out, bufA, E);
  rrelu_kernel<<<elemBlocks, 256, 0, stream>>>(bufA, out, (int)NH);
}
